// ContinuousNormalizingFlow_15006615732821
// MI455X (gfx1250) — matrix operand order repaired and checked by reading, not compiled, not run on hardware
//
#include <hip/hip_runtime.h>

typedef __attribute__((ext_vector_type(16))) _Float16 v16h;
typedef __attribute__((ext_vector_type(8)))  _Float16 v8h;
typedef __attribute__((ext_vector_type(4)))  _Float16 v4h;
typedef __attribute__((ext_vector_type(8)))  float    v8f;

#define DIM   64
#define HID   256
#define MT    32      // batch rows per workgroup
#define NT    512     // threads per workgroup (16 waves)
#define NSTEP 100
#define DTf   0.01f

// sched_group_barrier masks
#define SG_WMMA   0x008
#define SG_DSREAD 0x100
#define SGB(m, n) __builtin_amdgcn_sched_group_barrier((m), (n), 0)

__device__ __forceinline__ float tanh_fast(float x) {
  // tanh(x) = 1 - 2/(exp(2x)+1);  exp(2x) = exp2(2x*log2(e))
  float e = __builtin_amdgcn_exp2f(x * 2.8853900817779268f);
  return 1.0f - 2.0f * __builtin_amdgcn_rcpf(e + 1.0f);
}

// A-fragment (16x32 f16, row-major source in LDS), per ISA 16-bit A layout:
// lane = M (mod 16); elems 0..7 = K kBase+half*8+0..7 ; elems 8..15 = K kBase+16+half*8+0..7
__device__ __forceinline__ v16h load_a(const _Float16* base, int ld, int mBase, int kBase, int lane) {
  const int h = lane >> 4;
  const _Float16* p = base + (mBase + (lane & 15)) * ld + kBase + h * 8;
  v8h lo = *(const v8h*)p;        // ds_load_b128
  v8h hi = *(const v8h*)(p + 16); // ds_load_b128
  return __builtin_shufflevector(lo, hi, 0,1,2,3,4,5,6,7,8,9,10,11,12,13,14,15);
}

// B-fragment (32x16 f16) from row-major f32 weights in global (one-time).
// lane = N (mod 16); elems 0..7 = K kBase+half*8+0..7 ; elems 8..15 = K kBase+16+half*8+0..7
__device__ __forceinline__ v16h load_b_global(const float* W, int ldn, int kBase, int nBase, int lane) {
  const int h  = lane >> 4;
  const int n  = nBase + (lane & 15);
  const int k0 = kBase + h * 8;
  v16h b;
#pragma unroll
  for (int e = 0; e < 16; ++e) b[e] = (_Float16)W[(size_t)(k0 + (e & 7) + ((e >> 3) << 4)) * ldn + n];
  return b;
}

#define WMMA_F16(A, B, C) \
  __builtin_amdgcn_wmma_f32_16x16x32_f16(false, (A), false, (B), (short)0, (C), false, false)

__global__ __launch_bounds__(NT) void cnf_rk4_kernel(
    const float* __restrict__ z0, const float* __restrict__ W1, const float* __restrict__ b1,
    const float* __restrict__ W2, const float* __restrict__ b2,
    const float* __restrict__ W3, const float* __restrict__ b3,
    float* __restrict__ out)
{
  __shared__ float    Zcur[MT*DIM];   // current state (f32)
  __shared__ float    Zacc[MT*DIM];   // RK4 accumulator (f32)
  __shared__ float    Yb  [MT*DIM];   // f(z) output accumulator (f32, ds_add_f32 target)
  __shared__ _Float16 A1  [MT*DIM];   // layer-1 input (f16)
  __shared__ _Float16 A2  [MT*HID];   // layer-2 input (f16)
  __shared__ _Float16 A3  [MT*HID];   // layer-3 input (f16)

  const int tid  = threadIdx.x;
  const int lane = tid & 31;
  const int w    = tid >> 5;     // wave id 0..15
  const int h    = lane >> 4;
  const int ln   = lane & 15;

  const int nB = w * 16;         // hidden-column slice owned by this wave
  const int c3 = (w & 3) * 16;   // layer-3 output column block
  const int q3 = (w >> 2) * 64;  // layer-3 K block (K-split x4)

  // ---- one-time: weights -> registers (96 VGPRs of B-fragments) ----
  v16h B1f[2], B2f[8], B3f[2];
#pragma unroll
  for (int kb = 0; kb < 2; ++kb) B1f[kb] = load_b_global(W1, HID, kb*32, nB, lane);
#pragma unroll
  for (int kb = 0; kb < 8; ++kb) B2f[kb] = load_b_global(W2, HID, kb*32, nB, lane);
#pragma unroll
  for (int kb = 0; kb < 2; ++kb) B3f[kb] = load_b_global(W3, DIM, q3 + kb*32, c3, lane);
  const float b1v = b1[nB + ln];
  const float b2v = b2[nB + ln];
  const float b3v = (w < 4) ? b3[c3 + ln] : 0.0f;   // bias added by K-block-0 waves only

  // ---- one-time: load z tile, init RK4 state ----
  {
    float4 z = ((const float4*)z0)[(size_t)blockIdx.x * (MT*DIM/4) + tid];
    ((float4*)Zcur)[tid] = z;
    ((float4*)Zacc)[tid] = z;
    v4h e; e[0]=(_Float16)z.x; e[1]=(_Float16)z.y; e[2]=(_Float16)z.z; e[3]=(_Float16)z.w;
    *(v4h*)(A1 + 4*tid) = e;
    ((float4*)Yb)[tid] = make_float4(0.f,0.f,0.f,0.f);
  }
  __syncthreads();

  for (int step = 0; step < NSTEP; ++step) {
#pragma unroll
    for (int s = 0; s < 4; ++s) {
      // ---- layer 1: A1(32x64) x W1slice -> tanh -> A2 cols [nB,nB+16) ----
      {
        v8f c0 = {}, c1 = {};
#pragma unroll
        for (int kb = 0; kb < 2; ++kb) {
          v16h a0 = load_a(A1, DIM,  0, kb*32, lane);
          v16h a1 = load_a(A1, DIM, 16, kb*32, lane);
          c0 = WMMA_F16(a0, B1f[kb], c0);
          c1 = WMMA_F16(a1, B1f[kb], c1);
        }
        // force: all 8 ds_reads in flight first, then the 4 WMMAs (graduated waits)
        SGB(SG_DSREAD, 8);
        SGB(SG_WMMA, 4);
#pragma unroll
        for (int r = 0; r < 8; ++r)
          A2[(h*8 + r) * HID + nB + ln] = (_Float16)tanh_fast(c0[r] + b1v);
#pragma unroll
        for (int r = 0; r < 8; ++r)
          A2[(16 + h*8 + r) * HID + nB + ln] = (_Float16)tanh_fast(c1[r] + b1v);
      }
      __syncthreads();

      // ---- layer 2: A2(32x256) x W2slice -> tanh -> A3 ----
      {
        v8f c0 = {}, c1 = {};
#pragma unroll
        for (int kb = 0; kb < 8; ++kb) {
          v16h a0 = load_a(A2, HID,  0, kb*32, lane);
          v16h a1 = load_a(A2, HID, 16, kb*32, lane);
          c0 = WMMA_F16(a0, B2f[kb], c0);
          c1 = WMMA_F16(a1, B2f[kb], c1);
        }
        // pipelined schedule: DS(8) | {WMMA(2) DS(4)} x6 | WMMA(4)
        SGB(SG_DSREAD, 8);
#pragma unroll
        for (int i = 0; i < 6; ++i) {
          SGB(SG_WMMA, 2);
          SGB(SG_DSREAD, 4);
        }
        SGB(SG_WMMA, 4);
#pragma unroll
        for (int r = 0; r < 8; ++r)
          A3[(h*8 + r) * HID + nB + ln] = (_Float16)tanh_fast(c0[r] + b2v);
#pragma unroll
        for (int r = 0; r < 8; ++r)
          A3[(16 + h*8 + r) * HID + nB + ln] = (_Float16)tanh_fast(c1[r] + b2v);
      }
      __syncthreads();

      // ---- layer 3: A3(32x256) x W3 (K-split x4), ds_add_f32 reduce into Yb ----
      {
        v8f c0 = {}, c1 = {};
#pragma unroll
        for (int kb = 0; kb < 2; ++kb) {
          v16h a0 = load_a(A3, HID,  0, q3 + kb*32, lane);
          v16h a1 = load_a(A3, HID, 16, q3 + kb*32, lane);
          c0 = WMMA_F16(a0, B3f[kb], c0);
          c1 = WMMA_F16(a1, B3f[kb], c1);
        }
        SGB(SG_DSREAD, 8);
        SGB(SG_WMMA, 4);
#pragma unroll
        for (int r = 0; r < 8; ++r)
          atomicAdd(&Yb[(h*8 + r) * DIM + c3 + ln], c0[r] + b3v);
#pragma unroll
        for (int r = 0; r < 8; ++r)
          atomicAdd(&Yb[(16 + h*8 + r) * DIM + c3 + ln], c1[r] + b3v);
      }
      __syncthreads();

      // ---- RK4 element-wise combine; produce next eval's A1; zero Yb ----
      {
        const float ca = (s == 0 || s == 3) ? (DTf/6.0f) : (DTf/3.0f);
        float4 k  = ((float4*)Yb)[tid];
        float4 za = ((float4*)Zacc)[tid];
        za.x += ca*k.x; za.y += ca*k.y; za.z += ca*k.z; za.w += ca*k.w;
        ((float4*)Zacc)[tid] = za;
        v4h e;
        if (s < 3) {
          const float ce = (s == 2) ? DTf : (0.5f*DTf);
          float4 zc = ((float4*)Zcur)[tid];
          e[0]=(_Float16)(zc.x + ce*k.x);
          e[1]=(_Float16)(zc.y + ce*k.y);
          e[2]=(_Float16)(zc.z + ce*k.z);
          e[3]=(_Float16)(zc.w + ce*k.w);
        } else {
          ((float4*)Zcur)[tid] = za;   // commit step; Zacc already == new Zcur
          e[0]=(_Float16)za.x; e[1]=(_Float16)za.y; e[2]=(_Float16)za.z; e[3]=(_Float16)za.w;
        }
        *(v4h*)(A1 + 4*tid) = e;
        ((float4*)Yb)[tid] = make_float4(0.f,0.f,0.f,0.f);
      }
      __syncthreads();
    }
  }

  ((float4*)out)[(size_t)blockIdx.x * (MT*DIM/4) + tid] = ((float4*)Zcur)[tid];
}

extern "C" void kernel_launch(void* const* d_in, const int* in_sizes, int n_in,
                              void* d_out, int out_size, void* d_ws, size_t ws_size,
                              hipStream_t stream) {
  (void)in_sizes; (void)n_in; (void)out_size; (void)d_ws; (void)ws_size;
  const float* z0 = (const float*)d_in[0];
  const float* W1 = (const float*)d_in[1];
  const float* b1 = (const float*)d_in[2];
  const float* W2 = (const float*)d_in[3];
  const float* b2 = (const float*)d_in[4];
  const float* W3 = (const float*)d_in[5];
  const float* b3 = (const float*)d_in[6];
  float* out = (float*)d_out;

  dim3 grid(32768 / MT);   // 1024 workgroups, one 32-row batch tile each
  dim3 block(NT);          // 512 threads = 16 wave32
  cnf_rk4_kernel<<<grid, block, 0, stream>>>(z0, W1, b1, W2, b2, W3, b3, out);
}
